// TraceablePhimoeSparseMoeBlock_24137716203789
// MI455X (gfx1250) — matrix operand order repaired and checked by reading, not compiled, not run on hardware
//
#include <hip/hip_runtime.h>

typedef __attribute__((ext_vector_type(16))) __bf16 v16bf;
typedef __attribute__((ext_vector_type(8)))  float  v8f;

#define H_DIM   1024
#define F_DIM   2048
#define E_NUM   16
#define TILE_M  32              // two 16-row WMMA M-tiles per block
#define XLDS_LD (H_DIM + 8)     // bf16 elements per row, padded to dodge bank conflicts
#define H1_LD   (F_DIM + 8)

// ---- WMMA fragment builders (wave32; contiguous K order k = 16*(lane/16) + s, shared by A and B) ----

// A-matrix 16x32 bf16: lanes 0-15 = rows M=0..15 with K in [0..15];
// lanes 16-31 = same rows with K in [16..31].
__device__ __forceinline__ v16bf load_a_frag(const __bf16* src, int ld, int k0, int lane) {
  const int m = lane & 15, g = lane >> 4;
  const __bf16* row = src + m * ld + k0 + g * 16;
  v16bf a;
#pragma unroll
  for (int i = 0; i < 8; ++i) a[i] = row[i];
#pragma unroll
  for (int i = 0; i < 8; ++i) a[8 + i] = row[8 + i];
  return a;
}

// B-matrix 32x16 bf16 with B[k][n] = W[n][k] (W row-major [N][K], f32 source):
// lanes 0-15 -> N=lane, K=0..15; lanes 16-31 -> N=lane-16, K=16..31.
__device__ __forceinline__ v16bf load_b_frag_f32(const float* W, int ktot, int n0, int k0, int lane) {
  const int n = lane & 15, g = lane >> 4;
  const float* p = W + (size_t)(n0 + n) * ktot + (k0 + g * 16);
  v16bf b;
#pragma unroll
  for (int i = 0; i < 16; ++i) b[i] = (__bf16)p[i];
  return b;
}

__device__ __forceinline__ float fast_silu(float v) {
  return v * __builtin_amdgcn_rcpf(1.0f + __expf(-v));  // v_exp + v_rcp, no IEEE div
}

// ---- Kernel 1: zero output + per-expert counts ----
__global__ void moe_init_kernel(float* __restrict__ out, int n, int* __restrict__ counts) {
  int i = blockIdx.x * blockDim.x + threadIdx.x;
  if (i < n) out[i] = 0.0f;
  if (i < E_NUM) counts[i] = 0;
}

// ---- Kernel 2: router (logits -> softmax -> top2 -> renorm -> scatter) ----
__global__ void moe_router_kernel(const float* __restrict__ x, const float* __restrict__ Wg,
                                  int* __restrict__ counts, int* __restrict__ toklist,
                                  float* __restrict__ wlist, int T) {
  int t = blockIdx.x * blockDim.x + threadIdx.x;
  if (t >= T) return;
  float acc[E_NUM];
#pragma unroll
  for (int e = 0; e < E_NUM; ++e) acc[e] = 0.0f;
  const float* xt = x + (size_t)t * H_DIM;
  for (int h = 0; h < H_DIM; ++h) {
    float xv = xt[h];
#pragma unroll
    for (int e = 0; e < E_NUM; ++e) acc[e] += xv * Wg[e * H_DIM + h];  // uniform -> scalar loads
  }
  float mx = acc[0];
#pragma unroll
  for (int e = 1; e < E_NUM; ++e) mx = fmaxf(mx, acc[e]);
  // top-2 on logits (softmax is monotone); ties -> lowest index, as jax top_k
  int i0 = 0; float l0 = acc[0];
#pragma unroll
  for (int e = 1; e < E_NUM; ++e) if (acc[e] > l0) { l0 = acc[e]; i0 = e; }
  int i1 = -1; float l1 = -1e30f;
#pragma unroll
  for (int e = 0; e < E_NUM; ++e) if (e != i0 && acc[e] > l1) { l1 = acc[e]; i1 = e; }
  float e0 = __expf(l0 - mx), e1 = __expf(l1 - mx);
  float inv = 1.0f / (e0 + e1);
  float w0 = e0 * inv, w1 = e1 * inv;
  int p0 = atomicAdd(&counts[i0], 1);
  toklist[i0 * T + p0] = t;  wlist[i0 * T + p0] = w0;
  int p1 = atomicAdd(&counts[i1], 1);
  toklist[i1 * T + p1] = t;  wlist[i1 * T + p1] = w1;
}

// ---- Kernel 3: fused per-expert FFN over routed 32-token tiles ----
// Block = (expert, 32-token tile), 256 threads = 8 waves splitting N.
// Each wave runs 4 independent WMMA chains (2 M-tiles x 2 N-tiles); every
// converted B fragment feeds both M-tiles, halving cvt cost per WMMA.
__global__ void __launch_bounds__(256, 1)
moe_ffn_kernel(const float* __restrict__ x, const float* __restrict__ W1,
               const float* __restrict__ W2, const int* __restrict__ counts,
               const int* __restrict__ toklist, const float* __restrict__ wlist,
               float* __restrict__ out, int T) {
  __shared__ __bf16 x_lds[TILE_M * XLDS_LD];   // 66 KB
  __shared__ __bf16 h1_lds[TILE_M * H1_LD];    // 132 KB
  __shared__ int   s_tok[TILE_M];
  __shared__ float s_w[TILE_M];

  const int e    = blockIdx.y;
  const int tile = blockIdx.x;
  const int cnt  = counts[e];
  if (tile * TILE_M >= cnt) return;   // tile beyond this expert's token list

  const int tid = threadIdx.x;
  if (tid < TILE_M) {
    int idx = tile * TILE_M + tid;
    if (idx < cnt) { s_tok[tid] = toklist[e * T + idx]; s_w[tid] = wlist[e * T + idx]; }
    else           { s_tok[tid] = -1;                   s_w[tid] = 0.0f; }
  }
  __syncthreads();

  // Stage x rows f32 -> bf16 into LDS: 8 threads per row, 128 elems each.
  {
    const int r  = tid >> 3;                 // 0..31
    const int c0 = (tid & 7) * (H_DIM / 8);  // 0..896
    int tok = s_tok[r];
    __bf16* dst = &x_lds[r * XLDS_LD + c0];
    if (tok >= 0) {
      const float* src = x + (size_t)tok * H_DIM + c0;
#pragma unroll 8
      for (int i = 0; i < H_DIM / 8; ++i) dst[i] = (__bf16)src[i];
    } else {
#pragma unroll 8
      for (int i = 0; i < H_DIM / 8; ++i) dst[i] = (__bf16)0.0f;
    }
  }
  __syncthreads();

  const int wave = tid >> 5;   // 0..7
  const int lane = tid & 31;
  const int g = lane >> 4;
  const int n = lane & 15;

  // ---- GEMM1: h1 = silu(x @ W1[e]^T), N=F_DIM, K=H_DIM ----
  const float* W1e = W1 + (size_t)e * F_DIM * H_DIM;
  for (int p = wave; p < F_DIM / 32; p += 8) {
    const int n0 = p * 32;            // two adjacent 16-wide N tiles: n0, n0+16
    v8f acc00 = {}, acc01 = {};       // M-tile 0 x {N0, N1}
    v8f acc10 = {}, acc11 = {};       // M-tile 1 x {N0, N1}
    for (int k0 = 0; k0 < H_DIM; k0 += 32) {
      v16bf a0 = load_a_frag(x_lds,                      XLDS_LD, k0, lane);
      v16bf a1 = load_a_frag(x_lds + 16 * XLDS_LD,       XLDS_LD, k0, lane);
      v16bf b0 = load_b_frag_f32(W1e, H_DIM, n0,      k0, lane);
      v16bf b1 = load_b_frag_f32(W1e, H_DIM, n0 + 16, k0, lane);
      acc00 = __builtin_amdgcn_wmma_f32_16x16x32_bf16(false, a0, false, b0, (short)0, acc00, false, false);
      acc10 = __builtin_amdgcn_wmma_f32_16x16x32_bf16(false, a1, false, b0, (short)0, acc10, false, false);
      acc01 = __builtin_amdgcn_wmma_f32_16x16x32_bf16(false, a0, false, b1, (short)0, acc01, false, false);
      acc11 = __builtin_amdgcn_wmma_f32_16x16x32_bf16(false, a1, false, b1, (short)0, acc11, false, false);
    }
#pragma unroll
    for (int r = 0; r < 8; ++r) {
      const int m0 = g * 8 + r;
      const int m1 = 16 + m0;
      h1_lds[m0 * H1_LD + n0 + n]      = (__bf16)fast_silu(acc00[r]);
      h1_lds[m0 * H1_LD + n0 + 16 + n] = (__bf16)fast_silu(acc01[r]);
      h1_lds[m1 * H1_LD + n0 + n]      = (__bf16)fast_silu(acc10[r]);
      h1_lds[m1 * H1_LD + n0 + 16 + n] = (__bf16)fast_silu(acc11[r]);
    }
  }
  __syncthreads();

  // ---- GEMM2: out_tile = (h1 @ W2[e]^T) * w, N=H_DIM, K=F_DIM ----
  const float* W2e = W2 + (size_t)e * H_DIM * F_DIM;
  for (int p = wave; p < H_DIM / 32; p += 8) {
    const int n0 = p * 32;
    v8f acc00 = {}, acc01 = {};
    v8f acc10 = {}, acc11 = {};
    for (int k0 = 0; k0 < F_DIM; k0 += 32) {
      v16bf a0 = load_a_frag(h1_lds,                 H1_LD, k0, lane);
      v16bf a1 = load_a_frag(h1_lds + 16 * H1_LD,    H1_LD, k0, lane);
      v16bf b0 = load_b_frag_f32(W2e, F_DIM, n0,      k0, lane);
      v16bf b1 = load_b_frag_f32(W2e, F_DIM, n0 + 16, k0, lane);
      acc00 = __builtin_amdgcn_wmma_f32_16x16x32_bf16(false, a0, false, b0, (short)0, acc00, false, false);
      acc10 = __builtin_amdgcn_wmma_f32_16x16x32_bf16(false, a1, false, b0, (short)0, acc10, false, false);
      acc01 = __builtin_amdgcn_wmma_f32_16x16x32_bf16(false, a0, false, b1, (short)0, acc01, false, false);
      acc11 = __builtin_amdgcn_wmma_f32_16x16x32_bf16(false, a1, false, b1, (short)0, acc11, false, false);
    }
#pragma unroll
    for (int r = 0; r < 8; ++r) {
      const int m0 = g * 8 + r;
      const int m1 = 16 + m0;
      const int t0 = s_tok[m0];
      const int t1 = s_tok[m1];
      if (t0 >= 0) {
        float* dst = &out[(size_t)t0 * H_DIM + n0 + n];
        atomicAdd(dst,      acc00[r] * s_w[m0]);
        atomicAdd(dst + 16, acc01[r] * s_w[m0]);
      }
      if (t1 >= 0) {
        float* dst = &out[(size_t)t1 * H_DIM + n0 + n];
        atomicAdd(dst,      acc10[r] * s_w[m1]);
        atomicAdd(dst + 16, acc11[r] * s_w[m1]);
      }
    }
  }
}

extern "C" void kernel_launch(void* const* d_in, const int* in_sizes, int n_in,
                              void* d_out, int out_size, void* d_ws, size_t ws_size,
                              hipStream_t stream) {
  const float* x  = (const float*)d_in[0];   // [T, H]
  const float* Wg = (const float*)d_in[1];   // [E, H]
  const float* W1 = (const float*)d_in[2];   // [E, F, H]
  const float* W2 = (const float*)d_in[3];   // [E, H, F]
  float* out = (float*)d_out;

  const int T = in_sizes[0] / H_DIM;

  // Workspace layout: counts | toklist[E][T] | wlist[E][T]
  char* ws = (char*)d_ws;
  int*   counts  = (int*)ws;
  int*   toklist = (int*)(ws + 256);
  float* wlist   = (float*)(ws + 256 + (size_t)E_NUM * T * sizeof(int));

  const int n_out = T * H_DIM;
  moe_init_kernel<<<(n_out + 255) / 256, 256, 0, stream>>>(out, n_out, counts);
  moe_router_kernel<<<(T + 255) / 256, 256, 0, stream>>>(x, Wg, counts, toklist, wlist, T);

  dim3 grid((T + TILE_M - 1) / TILE_M, E_NUM);
  moe_ffn_kernel<<<grid, 256, 0, stream>>>(x, W1, W2, counts, toklist, wlist, out, T);
}
